// Attention_39436389712179
// MI455X (gfx1250) — matrix operand order repaired and checked by reading, not compiled, not run on hardware
//
#include <hip/hip_runtime.h>
#include <hip/hip_bf16.h>

#define DEV __device__ __forceinline__

typedef unsigned short US;
typedef unsigned int   U32;
typedef __attribute__((ext_vector_type(2)))  unsigned int u32x2;
typedef __attribute__((ext_vector_type(4)))  unsigned int u32x4;
typedef __attribute__((ext_vector_type(4)))  float        f32x4;
typedef __attribute__((ext_vector_type(16))) __bf16       v16bf;
typedef __attribute__((ext_vector_type(8)))  float        v8f;

union Frag { v16bf bf; u32x4 q[2]; };

DEV US f2bf(float f) {
    U32 u = __builtin_bit_cast(U32, f);
    return (US)((u + 0x7FFFu + ((u >> 16) & 1u)) >> 16);   // RNE f32->bf16
}
DEV U32 pk2(float a, float b) { return (U32)f2bf(a) | ((U32)f2bf(b) << 16); }
DEV v8f vzero() { v8f z; __builtin_memset(&z, 0, sizeof(z)); return z; }

#define WMMA_BF16(A_, B_, C_) \
    __builtin_amdgcn_wmma_f32_16x16x32_bf16(false, (A_), false, (B_), (short)0, (C_), false, false)

// CDNA5 async global->LDS copy (ASYNCcnt-tracked, no VGPR round trip).
// Generic pointer to __shared__ truncated to 32 bits == wave-relative LDS addr.
// Chunk displacement folded into the VGLOBAL 24-bit immediate offset field.
DEV U32 lds_addr(const void* p) { return (U32)(uintptr_t)p; }
template <int OFF>
DEV void async_b128(U32 lds, const US* g) {
    asm volatile("global_load_async_to_lds_b128 %0, %1, off offset:%2"
                 :: "v"(lds), "v"((unsigned long long)(uintptr_t)g), "i"(OFF)
                 : "memory");
}
DEV void wait_async0() { asm volatile("s_wait_asynccnt 0x0" ::: "memory"); }
DEV void wait_ds0()    { asm volatile("s_wait_dscnt 0x0" ::: "memory"); }

// ---------------------------------------------------------------------------
// fp32 -> bf16 elementwise (8 elems/thread)
// ---------------------------------------------------------------------------
__global__ __launch_bounds__(256)
void tobf16(const float* __restrict__ in, US* __restrict__ out) {
    const size_t base = ((size_t)blockIdx.x * 256 + threadIdx.x) * 8;
    f32x4 a0 = *(const f32x4*)(in + base);
    f32x4 a1 = *(const f32x4*)(in + base + 4);
    u32x4 pv = { pk2(a0[0], a0[1]), pk2(a0[2], a0[3]), pk2(a1[0], a1[1]), pk2(a1[2], a1[3]) };
    *(u32x4*)(out + base) = pv;
}

// ---------------------------------------------------------------------------
// W[K][N] fp32 -> Wt[N][K] bf16, 32x32 LDS-tiled transpose (one-shot cost).
// ---------------------------------------------------------------------------
__global__ __launch_bounds__(256)
void wtrans(const float* __restrict__ in, US* __restrict__ out, int K, int N) {
    __shared__ float tile[32][33];
    const int k0 = blockIdx.y * 32, n0 = blockIdx.x * 32;
    const int t = threadIdx.x;
    {
        const int kl = t >> 3, nl4 = (t & 7) * 4;
        f32x4 v4 = *(const f32x4*)(in + (size_t)(k0 + kl) * N + n0 + nl4);
#pragma unroll
        for (int e = 0; e < 4; ++e) tile[kl][nl4 + e] = v4[e];
    }
    __syncthreads();
    {
        const int nl = t >> 3, kl4 = (t & 7) * 4;
        US o[4];
#pragma unroll
        for (int e = 0; e < 4; ++e) o[e] = f2bf(tile[kl4 + e][nl]);
        u32x2 pv = { (U32)o[0] | ((U32)o[1] << 16), (U32)o[2] | ((U32)o[3] << 16) };
        *(u32x2*)(out + (size_t)(n0 + nl) * K + k0 + kl4) = pv;
    }
}

// ---------------------------------------------------------------------------
// GEMM: C[M,N] fp32 = A[M,K](bf16) * Wt[N,K](bf16, pre-transposed).
// 8 waves, block tile 128x128, wave tile 32x64, K-step 32.
// Both operand tiles staged by async global->LDS b128 copies, double-buffered:
// issue(next) right after barrier, WMMA on current, wait at top of next iter.
// ---------------------------------------------------------------------------
__global__ __launch_bounds__(256)
void gemm_bf16t(const US* __restrict__ A, const US* __restrict__ Wt,
                float* __restrict__ C, int M, int N, int K) {
    __shared__ __align__(16) US sA[2][128 * 40];   // [m][k], row stride 40 (16B aligned)
    __shared__ __align__(16) US sB[2][128 * 40];   // [n][k]

    const int t = threadIdx.x, lane = t & 31, w = t >> 5;
    const int wm = w >> 1, wn = w & 1;
    const int la = lane & 15, lh = lane >> 4;
    const int m0 = blockIdx.y * 128, n0 = blockIdx.x * 128;

    // staging map: thread t copies 2 b128 chunks of the A tile and 2 of the B tile
    const int ar = t >> 1, as_ = (t & 1) * 16;          // row 0..127, ushort offset 0/16
    const US* gA = A + (size_t)(m0 + ar) * K + as_;
    const US* gB = Wt + (size_t)(n0 + ar) * K + as_;
    U32 lA[2], lB[2];
#pragma unroll
    for (int bfi = 0; bfi < 2; ++bfi) {
        lA[bfi] = lds_addr(&sA[bfi][ar * 40 + as_]);
        lB[bfi] = lds_addr(&sB[bfi][ar * 40 + as_]);
    }

    v8f acc[2][4];
#pragma unroll
    for (int i = 0; i < 2; ++i)
#pragma unroll
        for (int j = 0; j < 4; ++j) acc[i][j] = vzero();

    const int nk = K / 32;
    // prologue: stage tile 0
    async_b128<0 >(lA[0], gA);   async_b128<16>(lA[0] + 16, gA);
    async_b128<0 >(lB[0], gB);   async_b128<16>(lB[0] + 16, gB);

    for (int kt = 0; kt < nk; ++kt) {
        const int cur = kt & 1;
        wait_async0();
        __syncthreads();
        if (kt + 1 < nk) {
            const int nxt = cur ^ 1;
            const US* pA = gA + (size_t)(kt + 1) * 32;
            const US* pB = gB + (size_t)(kt + 1) * 32;
            async_b128<0 >(lA[nxt], pA);   async_b128<16>(lA[nxt] + 16, pA);
            async_b128<0 >(lB[nxt], pB);   async_b128<16>(lB[nxt] + 16, pB);
        }

        Frag a[2], b[4];
#pragma unroll
        for (int mt = 0; mt < 2; ++mt) {
            const int off = (wm * 32 + mt * 16 + la) * 40 + lh * 8;
            a[mt].q[0] = *(const u32x4*)(sA[cur] + off);
            a[mt].q[1] = *(const u32x4*)(sA[cur] + off + 16);
        }
#pragma unroll
        for (int nt = 0; nt < 4; ++nt) {
            const int off = (wn * 64 + nt * 16 + la) * 40 + lh * 8;
            b[nt].q[0] = *(const u32x4*)(sB[cur] + off);
            b[nt].q[1] = *(const u32x4*)(sB[cur] + off + 16);
        }
#pragma unroll
        for (int mt = 0; mt < 2; ++mt)
#pragma unroll
            for (int nt = 0; nt < 4; ++nt)
                acc[mt][nt] = WMMA_BF16(a[mt].bf, b[nt].bf, acc[mt][nt]);
    }

#pragma unroll
    for (int mt = 0; mt < 2; ++mt)
#pragma unroll
        for (int nt = 0; nt < 4; ++nt) {
            const int col = n0 + wn * 64 + nt * 16 + la;
#pragma unroll
            for (int j = 0; j < 8; ++j) {
                const int row = m0 + wm * 32 + mt * 16 + lh * 8 + j;
                C[(size_t)row * N + col] = acc[mt][nt][j];
            }
        }
}

// ---------------------------------------------------------------------------
// LayerNorm (+optional RoPE) over 2048-wide rows; writes bf16 [b][h][rows][128].
// For Q: softmax scale * log2(e) folded in so attention uses exp2.
// ---------------------------------------------------------------------------
template <bool ROPE>
__global__ __launch_bounds__(256)
void ln_rope(const float* __restrict__ in, const float* __restrict__ w,
             const float* __restrict__ bias, const float* __restrict__ freqs,
             float scale, float eps, US* __restrict__ out, int rows) {
    const int r = blockIdx.x;               // 0..B*rows-1
    const int b = r / rows, s = r - b * rows;
    const int t = threadIdx.x, lane = t & 31, wv = t >> 5;
    const int base = t * 8;

    const float* rowp = in + (size_t)r * 2048;
    float v[8];
    {
        f32x4 a0 = *(const f32x4*)(rowp + base);
        f32x4 a1 = *(const f32x4*)(rowp + base + 4);
#pragma unroll
        for (int e = 0; e < 4; ++e) { v[e] = a0[e]; v[4 + e] = a1[e]; }
    }
    float sm = 0.f, sq = 0.f;
#pragma unroll
    for (int e = 0; e < 8; ++e) { sm += v[e]; sq += v[e] * v[e]; }
#pragma unroll
    for (int m = 16; m >= 1; m >>= 1) {
        sm += __shfl_xor(sm, m, 32);
        sq += __shfl_xor(sq, m, 32);
    }
    __shared__ float red[16];
    if (lane == 0) { red[wv] = sm; red[8 + wv] = sq; }
    __syncthreads();
    float ts = 0.f, ts2 = 0.f;
#pragma unroll
    for (int i = 0; i < 8; ++i) { ts += red[i]; ts2 += red[8 + i]; }
    const float mu = ts * (1.0f / 2048.0f);
    const float var = ts2 * (1.0f / 2048.0f) - mu * mu;
    const float rs = rsqrtf(var + eps);

#pragma unroll
    for (int e = 0; e < 8; ++e) {
        const int d = base + e;
        v[e] = ((v[e] - mu) * rs * w[d] + bias[d]) * scale;
    }
    if (ROPE) {
#pragma unroll
        for (int e = 0; e < 8; e += 2) {
            const int d = base + e;
            const int fi = (s * 64 + ((d & 127) >> 1)) * 2;
            const float c = freqs[fi], sn = freqs[fi + 1];
            const float a0 = v[e], a1 = v[e + 1];
            v[e]     = a0 * c - a1 * sn;
            v[e + 1] = a0 * sn + a1 * c;
        }
    }
    const int h = base >> 7, hd0 = base & 127;
    u32x4 pv = { pk2(v[0], v[1]), pk2(v[2], v[3]), pk2(v[4], v[5]), pk2(v[6], v[7]) };
    US* op = out + (((size_t)b * 16 + h) * rows + s) * 128 + hd0;
    *(u32x4*)op = pv;
}

// ---------------------------------------------------------------------------
// V convert fp32 [b][s][h*128+hd] -> bf16 transposed [b][h][hd][rows]
// ---------------------------------------------------------------------------
__global__ __launch_bounds__(256)
void vconv(const float* __restrict__ in, US* __restrict__ out, int rows) {
    const size_t i = (size_t)blockIdx.x * 256 + threadIdx.x;
    const int d = (int)(i & 2047), hd = d & 127, h = d >> 7;
    const size_t bs = i >> 11;
    const int s = (int)(bs % rows), b = (int)(bs / rows);
    out[(((size_t)b * 16 + h) * 128 + hd) * rows + s] = f2bf(in[i]);
}

// ---------------------------------------------------------------------------
// Flash attention. Q bf16 [b][h][Sq][128] (Q pre-scaled by scale*log2e),
// K bf16 [b][h][Tkv][128], Vt bf16 [b][h][128][Tkv]. O fp32 [b][Sq][2048].
// gate==nullptr: O = attn ; else O += tanh(gate[h]) * attn.
// K/Vt tiles staged with async global->LDS b128 copies (offsets as immediates).
// ---------------------------------------------------------------------------
__global__ __launch_bounds__(256)
void flash(const US* __restrict__ Q, const US* __restrict__ Kb,
           const US* __restrict__ Vt, float* __restrict__ O,
           const float* __restrict__ gate, int Sq, int Tkv) {
    __shared__ __align__(16) US sK[64 * 136];     // [key][hd]
    __shared__ __align__(16) US sV[128 * 72];     // [hd][key]
    __shared__ __align__(16) US sP[8 * 16 * 72];  // per-wave P tiles [row][key]

    const int b = blockIdx.z, h = blockIdx.y;
    const int t = threadIdx.x, lane = t & 31, wv = t >> 5;
    const int la = lane & 15, lh = lane >> 4;
    const int q0 = blockIdx.x * 128 + wv * 16;
    const int wbase = wv * 16 * 72;

    // Q fragments resident in registers for the whole kernel
    Frag aq[4];
    {
        const US* qrow = Q + ((size_t)(b * 16 + h) * Sq + q0 + la) * 128;
#pragma unroll
        for (int kk = 0; kk < 4; ++kk) {
            const int off = kk * 32 + lh * 8;
            aq[kk].q[0] = *(const u32x4*)(qrow + off);
            aq[kk].q[1] = *(const u32x4*)(qrow + off + 16);
        }
    }

    v8f o[8];
#pragma unroll
    for (int i = 0; i < 8; ++i) o[i] = vzero();
    float mi[8], li[8];
#pragma unroll
    for (int j = 0; j < 8; ++j) { mi[j] = -1e30f; li[j] = 0.f; }

    const US* kbase = Kb + (size_t)(b * 16 + h) * Tkv * 128;
    const US* vbase = Vt + (size_t)(b * 16 + h) * 128 * Tkv;

    // async staging maps
    const int kr = t >> 2, kc = (t & 3) * 32;     // K: row 0..63, ushort col 0..96
    const int vr = t >> 1, vc = (t & 1) * 32;     // V: hd 0..127, ushort col 0/32
    const U32 lK = lds_addr(&sK[kr * 136 + kc]);
    const U32 lV = lds_addr(&sV[vr * 72 + vc]);
    const US* gk0 = kbase + (size_t)kr * 128 + kc;
    const US* gv0 = vbase + (size_t)vr * Tkv + vc;

    for (int kt = 0; kt < Tkv; kt += 64) {
        {
            const US* gk = gk0 + (size_t)kt * 128;
            const US* gv = gv0 + kt;
            async_b128<0 >(lK,      gk);  async_b128<0 >(lV,      gv);
            async_b128<16>(lK + 16, gk);  async_b128<16>(lV + 16, gv);
            async_b128<32>(lK + 32, gk);  async_b128<32>(lV + 32, gv);
            async_b128<48>(lK + 48, gk);  async_b128<48>(lV + 48, gv);
        }
        wait_async0();
        __syncthreads();

        // scores S = Q @ K^T (16 x 64 per wave), already in log2 domain
        v8f sc[4];
#pragma unroll
        for (int nt = 0; nt < 4; ++nt) {
            sc[nt] = vzero();
            const int ro = (nt * 16 + la) * 136;
#pragma unroll
            for (int kk = 0; kk < 4; ++kk) {
                Frag bk;
                const int off = ro + kk * 32 + lh * 8;
                bk.q[0] = *(const u32x4*)(sK + off);
                bk.q[1] = *(const u32x4*)(sK + off + 16);
                sc[nt] = WMMA_BF16(aq[kk].bf, bk.bf, sc[nt]);
            }
        }

        // online softmax (rows striped: row = j + 8*lh, 16 lanes per row)
#pragma unroll
        for (int j = 0; j < 8; ++j) {
            float s0 = sc[0][j], s1 = sc[1][j], s2 = sc[2][j], s3 = sc[3][j];
            float mx = fmaxf(fmaxf(s0, s1), fmaxf(s2, s3));
#pragma unroll
            for (int m = 8; m >= 1; m >>= 1) mx = fmaxf(mx, __shfl_xor(mx, m, 32));
            const float mn = fmaxf(mi[j], mx);
            const float al = exp2f(mi[j] - mn);
            const float p0 = exp2f(s0 - mn), p1 = exp2f(s1 - mn);
            const float p2 = exp2f(s2 - mn), p3 = exp2f(s3 - mn);
            float ls = p0 + p1 + p2 + p3;
#pragma unroll
            for (int m = 8; m >= 1; m >>= 1) ls += __shfl_xor(ls, m, 32);
            li[j] = li[j] * al + ls;
            mi[j] = mn;
#pragma unroll
            for (int ht = 0; ht < 8; ++ht) o[ht][j] *= al;
            const int rr = wbase + (j + 8 * lh) * 72 + la;
            sP[rr]      = f2bf(p0);
            sP[rr + 16] = f2bf(p1);
            sP[rr + 32] = f2bf(p2);
            sP[rr + 48] = f2bf(p3);
        }
        wait_ds0();   // wave-local P tile visible to all lanes

        // O += P @ V  (16 x 128 per wave)
        Frag ap[2];
#pragma unroll
        for (int kk2 = 0; kk2 < 2; ++kk2) {
            const int off = wbase + la * 72 + kk2 * 32 + lh * 8;
            ap[kk2].q[0] = *(const u32x4*)(sP + off);
            ap[kk2].q[1] = *(const u32x4*)(sP + off + 16);
        }
#pragma unroll
        for (int ht = 0; ht < 8; ++ht) {
            const int ro = (ht * 16 + la) * 72;
#pragma unroll
            for (int kk2 = 0; kk2 < 2; ++kk2) {
                Frag bv;
                const int off = ro + kk2 * 32 + lh * 8;
                bv.q[0] = *(const u32x4*)(sV + off);
                bv.q[1] = *(const u32x4*)(sV + off + 16);
                o[ht] = WMMA_BF16(ap[kk2].bf, bv.bf, o[ht]);
            }
        }
        __syncthreads();   // reads done before next tile's async writes
    }

    const float g = gate ? tanhf(gate[h]) : 1.0f;
#pragma unroll
    for (int j = 0; j < 8; ++j) {
        const float inv = 1.0f / li[j];
        const size_t rowoff = ((size_t)b * Sq + q0 + j + 8 * lh) * 2048 + h * 128;
#pragma unroll
        for (int ht = 0; ht < 8; ++ht) {
            const float val = o[ht][j] * inv;
            float* p = O + rowoff + ht * 16 + la;
            if (gate) *p += g * val; else *p = val;
        }
    }
}

// ---------------------------------------------------------------------------
extern "C" void kernel_launch(void* const* d_in, const int* in_sizes, int n_in,
                              void* d_out, int out_size, void* d_ws, size_t ws_size,
                              hipStream_t stream) {
    const float* x     = (const float*)d_in[0];
    const float* freqs = (const float*)d_in[2];
    const float* y     = (const float*)d_in[3];
    const float* wq    = (const float*)d_in[5];
    const float* wk    = (const float*)d_in[6];
    const float* wv    = (const float*)d_in[7];
    const float* wo    = (const float*)d_in[8];
    const float* wky   = (const float*)d_in[9];
    const float* wvy   = (const float*)d_in[10];
    const float* gate  = (const float*)d_in[11];
    const float* qn_w  = (const float*)d_in[12];
    const float* qn_b  = (const float*)d_in[13];
    const float* kn_w  = (const float*)d_in[14];
    const float* kn_b  = (const float*)d_in[15];
    const float* kyn_w = (const float*)d_in[16];
    const float* kyn_b = (const float*)d_in[17];
    float* out = (float*)d_out;

    char* ws = (char*)d_ws;
    const size_t SZ_XF = (size_t)4096 * 2048 * 4;   // fp32  [4096][2048]
    const size_t SZ_YF = (size_t)1024 * 2048 * 4;
    const size_t SZ_XB = (size_t)4096 * 2048 * 2;   // bf16  [4096][2048]
    const size_t SZ_YB = (size_t)1024 * 2048 * 2;
    const size_t SZ_WB = (size_t)2048 * 2048 * 2;   // bf16  [2048][2048]
    size_t off = 0;
    float* xq_f = (float*)(ws + off); off += SZ_XF;
    float* xk_f = (float*)(ws + off); off += SZ_XF;
    float* xv_f = (float*)(ws + off); off += SZ_XF;
    float* yk_f = (float*)(ws + off); off += SZ_YF;
    float* yv_f = (float*)(ws + off); off += SZ_YF;
    US* q_bf   = (US*)(ws + off); off += SZ_XB;
    US* k_bf   = (US*)(ws + off); off += SZ_XB;
    US* vt_bf  = (US*)(ws + off); off += SZ_XB;
    US* yk_bf  = (US*)(ws + off); off += SZ_YB;
    US* yvt_bf = (US*)(ws + off); off += SZ_YB;
    US* x_bf   = (US*)(ws + off); off += SZ_XB;
    US* y_bf   = (US*)(ws + off); off += SZ_YB;
    US* o_bf   = (US*)(ws + off); off += SZ_XB;
    US* wq_t   = (US*)(ws + off); off += SZ_WB;
    US* wk_t   = (US*)(ws + off); off += SZ_WB;
    US* wv_t   = (US*)(ws + off); off += SZ_WB;
    US* wo_t   = (US*)(ws + off); off += SZ_WB;
    US* wky_t  = (US*)(ws + off); off += SZ_WB;
    US* wvy_t  = (US*)(ws + off); off += SZ_WB;
    float* O = xq_f;   // xq_f is dead after ln_rope -> reuse for attention output

    const dim3 blk(256);
    const dim3 gwt(2048 / 32, 2048 / 32);

    // one-shot precision conversion + weight transposes (bandwidth trivial)
    tobf16<<<dim3(4096), blk, 0, stream>>>(x, x_bf);
    tobf16<<<dim3(1024), blk, 0, stream>>>(y, y_bf);
    wtrans<<<gwt, blk, 0, stream>>>(wq,  wq_t,  2048, 2048);
    wtrans<<<gwt, blk, 0, stream>>>(wk,  wk_t,  2048, 2048);
    wtrans<<<gwt, blk, 0, stream>>>(wv,  wv_t,  2048, 2048);
    wtrans<<<gwt, blk, 0, stream>>>(wo,  wo_t,  2048, 2048);
    wtrans<<<gwt, blk, 0, stream>>>(wky, wky_t, 2048, 2048);
    wtrans<<<gwt, blk, 0, stream>>>(wvy, wvy_t, 2048, 2048);

    const dim3 gx(2048 / 128, 4096 / 128);
    const dim3 gy(2048 / 128, 1024 / 128);
    gemm_bf16t<<<gx, blk, 0, stream>>>(x_bf, wq_t, xq_f, 4096, 2048, 2048);
    gemm_bf16t<<<gx, blk, 0, stream>>>(x_bf, wk_t, xk_f, 4096, 2048, 2048);
    gemm_bf16t<<<gx, blk, 0, stream>>>(x_bf, wv_t, xv_f, 4096, 2048, 2048);
    gemm_bf16t<<<gy, blk, 0, stream>>>(y_bf, wky_t, yk_f, 1024, 2048, 2048);
    gemm_bf16t<<<gy, blk, 0, stream>>>(y_bf, wvy_t, yv_f, 1024, 2048, 2048);

    // softmax scale folded into Q, times log2(e) for exp2-domain flash softmax
    const float QSCALE = 0.08838834764831845f * 1.4426950408889634f;
    ln_rope<true ><<<dim3(4096), blk, 0, stream>>>(xq_f, qn_w, qn_b, freqs, QSCALE, 1e-5f, q_bf, 2048);
    ln_rope<true ><<<dim3(4096), blk, 0, stream>>>(xk_f, kn_w, kn_b, freqs, 1.0f,   1e-5f, k_bf, 2048);
    ln_rope<false><<<dim3(1024), blk, 0, stream>>>(yk_f, kyn_w, kyn_b, freqs, 1.0f, 1e-6f, yk_bf, 512);
    vconv<<<dim3((4096 * 2048) / 256), blk, 0, stream>>>(xv_f, vt_bf, 2048);
    vconv<<<dim3((1024 * 2048) / 256), blk, 0, stream>>>(yv_f, yvt_bf, 512);

    const dim3 gf(2048 / 128, 16, 2);
    flash<<<gf, blk, 0, stream>>>(q_bf, k_bf,  vt_bf,  O, nullptr, 2048, 2048);
    flash<<<gf, blk, 0, stream>>>(q_bf, yk_bf, yvt_bf, O, gate,    2048, 512);

    tobf16<<<dim3(4096), blk, 0, stream>>>(O, o_bf);
    gemm_bf16t<<<gx, blk, 0, stream>>>(o_bf, wo_t, out, 4096, 2048, 2048);
}
